// Actor_54400055771818
// MI455X (gfx1250) — matrix operand order repaired and checked by reading, not compiled, not run on hardware
//
#include <hip/hip_runtime.h>
#include <stdint.h>

#define NNODES 100000
#define NEDGES 3200000
#define HID    32
#define SLOPE  0.01f
#define EPB    128   // edges per block (8 waves x 16-edge WMMA tiles)
#define BLOCK  256
#define NWAVE  8

typedef __bf16 bf16;
typedef __attribute__((ext_vector_type(16))) bf16  v16bf;
typedef __attribute__((ext_vector_type(8)))  bf16  v8bf;
typedef __attribute__((ext_vector_type(8)))  float v8f;

union AFrag { v16bf v; v8bf h[2]; };

__device__ inline unsigned bfbits(float f) {           // RNE float->bf16 bits
  unsigned u = __float_as_uint(f);
  return (u + 0x7FFFu + ((u >> 16) & 1u)) >> 16;
}
__device__ inline bf16 f2bf(float f) {
  unsigned short s = (unsigned short)bfbits(f);
  bf16 b; __builtin_memcpy(&b, &s, 2); return b;
}
__device__ inline unsigned pack2bf(float a, float b) { // two floats -> packed bf16x2
  return (bfbits(a) & 0xFFFFu) | (bfbits(b) << 16);
}
__device__ inline unsigned ordOf(float f) {            // monotonic float->uint
  unsigned u = __float_as_uint(f);
  return (u & 0x80000000u) ? ~u : (u | 0x80000000u);
}
__device__ inline float lrelu(float v) { return v > 0.f ? v : v * SLOPE; }

// ---------------------------------------------------------------------------
// Edge-MLP + atomic min-aggregate.  INF = per-node feature width (4 or 32),
// KPAD = concat width (2*INF+4) padded to a multiple of 32.
// ---------------------------------------------------------------------------
template<int INF, int KPAD>
__global__ __launch_bounds__(BLOCK)
void edge_conv_kernel(const float* __restrict__ feat,
                      const long long* __restrict__ ei,
                      const float* __restrict__ ea,
                      const float* __restrict__ W1, const float* __restrict__ b1,
                      const float* __restrict__ W2, const float* __restrict__ b2,
                      unsigned* __restrict__ agg, int nEdges)
{
  constexpr int KIN = 2 * INF + 4;
  constexpr int CH  = KPAD / 32;
  constexpr int KP2 = KPAD / 2;

  __shared__ int sTgt[EPB];
  __shared__ int sSrc[EPB];
  __shared__ __align__(16) bf16 sA[EPB * KPAD];          // staged concat rows
  __shared__ __align__(16) bf16 sMid[NWAVE * 16 * HID];  // layer-1 tile bounce

  const int tid  = threadIdx.x;
  const int lane = tid & 31;
  const int wv   = tid >> 5;
  const int n    = lane & 15;   // column within 16-wide N tile / A row within tile
  const int kh   = lane >> 4;   // k-half select
  const int e0   = blockIdx.x * EPB;

  // ---- edge endpoints (reference: src = row0, tgt = row1) ----
  if (tid < EPB) {
    int ge = e0 + tid; if (ge >= nEdges) ge = nEdges - 1;
    sSrc[tid] = (int)ei[ge];
    sTgt[tid] = (int)ei[(long long)NEDGES + ge];
  }
  __syncthreads();

  // ---- weight B-fragments into registers (lane = column n, K-run from kh) ----
  v16bf w1f[CH][2];
  v16bf w2f[2];
  #pragma unroll
  for (int c = 0; c < CH; ++c)
    #pragma unroll
    for (int nt = 0; nt < 2; ++nt)
      #pragma unroll
      for (int i = 0; i < 16; ++i) {
        int k = c * 32 + kh * 16 + i;
        w1f[c][nt][i] = f2bf(k < KIN ? W1[k * HID + nt * 16 + n] : 0.f);
      }
  #pragma unroll
  for (int nt = 0; nt < 2; ++nt)
    #pragma unroll
    for (int i = 0; i < 16; ++i)
      w2f[nt][i] = f2bf(W2[(kh * 16 + i) * HID + nt * 16 + n]);

  // ---- stage concat([h_tgt, h_src, edge_attr], pad) rows as bf16 in LDS ----
  // Pair-granular: float2 global loads, packed-bf16x2 b32 LDS stores.
  for (int p = tid; p < EPB * KP2; p += BLOCK) {
    int e  = p / KP2;
    int kp = p - e * KP2;
    int k  = kp * 2;
    int ge = e0 + e; if (ge >= nEdges) ge = nEdges - 1;
    float2 v = make_float2(0.f, 0.f);
    if      (k < INF)     v = *(const float2*)&feat[(long long)sTgt[e] * INF + k];
    else if (k < 2 * INF) v = *(const float2*)&feat[(long long)sSrc[e] * INF + (k - INF)];
    else if (k < KIN)     v = *(const float2*)&ea[(long long)ge * 4 + (k - 2 * INF)];
    *(unsigned*)&sA[e * KPAD + k] = pack2bf(v.x, v.y);
  }
  __syncthreads();

  // ---- layer 1: C += A(chunk) x W1(chunk) over K ----
  const int rowBase = wv * 16;
  v8f c1a = {};
  v8f c1b = {};
  #pragma unroll
  for (int c = 0; c < CH; ++c) {
    const bf16* rp = &sA[(rowBase + n) * KPAD + c * 32 + kh * 16];
    AFrag a;
    a.h[0] = *(const v8bf*)rp;           // K = c*32 + kh*16 .. +7
    a.h[1] = *(const v8bf*)(rp + 8);     // K = c*32 + kh*16 + 8 .. +15
    c1a = __builtin_amdgcn_wmma_f32_16x16x32_bf16(false, a.v, false, w1f[c][0],
                                                  (short)0, c1a, false, false);
    c1b = __builtin_amdgcn_wmma_f32_16x16x32_bf16(false, a.v, false, w1f[c][1],
                                                  (short)0, c1b, false, false);
  }

  // ---- bias + LeakyReLU, bounce tile through LDS to re-fragment as A ----
  const float bia = b1[n];
  const float bib = b1[16 + n];
  bf16* mid = &sMid[wv * 16 * HID];
  #pragma unroll
  for (int r = 0; r < 8; ++r) {
    int m = r + 8 * kh;
    mid[m * HID + n]      = f2bf(lrelu(c1a[r] + bia));
    mid[m * HID + 16 + n] = f2bf(lrelu(c1b[r] + bib));
  }
  __syncthreads();

  // ---- layer 2 (K = 32, one chunk) ----
  AFrag a2;
  {
    const bf16* rp = &mid[n * HID + kh * 16];
    a2.h[0] = *(const v8bf*)rp;
    a2.h[1] = *(const v8bf*)(rp + 8);
  }
  v8f c2a = {};
  v8f c2b = {};
  c2a = __builtin_amdgcn_wmma_f32_16x16x32_bf16(false, a2.v, false, w2f[0],
                                                (short)0, c2a, false, false);
  c2b = __builtin_amdgcn_wmma_f32_16x16x32_bf16(false, a2.v, false, w2f[1],
                                                (short)0, c2b, false, false);

  // ---- min-aggregate per target node (ordered-uint atomicMin) ----
  const float b2a = b2[n];
  const float b2b = b2[16 + n];
  #pragma unroll
  for (int r = 0; r < 8; ++r) {
    int m = r + 8 * kh;
    int ge = e0 + rowBase + m;
    if (ge < nEdges) {
      unsigned* dst = &agg[(long long)sTgt[rowBase + m] * HID];
      atomicMin(dst + n,      ordOf(c2a[r] + b2a));
      atomicMin(dst + 16 + n, ordOf(c2b[r] + b2b));
    }
  }
}

// ---------------------------------------------------------------------------
__global__ void fill_kernel(unsigned* __restrict__ p, int n) {
  int i = blockIdx.x * blockDim.x + threadIdx.x;
  if (i < n) p[i] = 0xFFFFFFFFu;   // "no edge" sentinel (> any encoded float)
}

__global__ void decode_kernel(const unsigned* __restrict__ agg,
                              float* __restrict__ h, int n) {
  int i = blockIdx.x * blockDim.x + threadIdx.x;
  if (i < n) {
    unsigned e = agg[i];
    float v = 0.f;
    if (e != 0xFFFFFFFFu) {
      unsigned u = (e & 0x80000000u) ? (e & 0x7FFFFFFFu) : ~e;
      v = lrelu(__uint_as_float(u));   // outer LeakyReLU; isolated nodes stay 0
    }
    h[i] = v;
  }
}

__global__ void head_kernel(const float* __restrict__ x, const float* __restrict__ h,
                            const float* __restrict__ W, const float* __restrict__ b,
                            float* __restrict__ out, int n) {
  int i = blockIdx.x * blockDim.x + threadIdx.x;
  if (i < n) {
    float acc = b[0];
    #pragma unroll
    for (int k = 0; k < 4; ++k)  acc += x[(long long)i * 4 + k] * W[k];
    #pragma unroll
    for (int k = 0; k < HID; ++k) acc += h[(long long)i * HID + k] * W[4 + k];
    out[i] = acc;
  }
}

// ---------------------------------------------------------------------------
extern "C" void kernel_launch(void* const* d_in, const int* in_sizes, int n_in,
                              void* d_out, int out_size, void* d_ws, size_t ws_size,
                              hipStream_t stream) {
  const float*     x  = (const float*)d_in[0];
  const long long* ei = (const long long*)d_in[1];
  const float*     ea = (const float*)d_in[2];

  unsigned* agg = (unsigned*)d_ws;                                   // N*32 u32
  float*    h   = (float*)((char*)d_ws + (size_t)NNODES * HID * 4);  // N*32 f32

  const int aggN   = NNODES * HID;
  const int fillGr = (aggN + 255) / 256;
  const int edgeGr = (NEDGES + EPB - 1) / EPB;

  // conv1 (input = x, INF=4, K=12 -> pad 32)
  fill_kernel<<<fillGr, 256, 0, stream>>>(agg, aggN);
  edge_conv_kernel<4, 32><<<edgeGr, BLOCK, 0, stream>>>(
      x, ei, ea,
      (const float*)d_in[3], (const float*)d_in[4],
      (const float*)d_in[5], (const float*)d_in[6], agg, NEDGES);
  decode_kernel<<<fillGr, 256, 0, stream>>>(agg, h, aggN);

  // conv2 (input = h, INF=32, K=68 -> pad 96)
  fill_kernel<<<fillGr, 256, 0, stream>>>(agg, aggN);
  edge_conv_kernel<32, 96><<<edgeGr, BLOCK, 0, stream>>>(
      h, ei, ea,
      (const float*)d_in[7], (const float*)d_in[8],
      (const float*)d_in[9], (const float*)d_in[10], agg, NEDGES);
  decode_kernel<<<fillGr, 256, 0, stream>>>(agg, h, aggN);

  // conv3
  fill_kernel<<<fillGr, 256, 0, stream>>>(agg, aggN);
  edge_conv_kernel<32, 96><<<edgeGr, BLOCK, 0, stream>>>(
      h, ei, ea,
      (const float*)d_in[11], (const float*)d_in[12],
      (const float*)d_in[13], (const float*)d_in[14], agg, NEDGES);
  decode_kernel<<<fillGr, 256, 0, stream>>>(agg, h, aggN);

  // head: concat([x, h]) @ head_W + head_b
  head_kernel<<<(NNODES + 255) / 256, 256, 0, stream>>>(
      x, h, (const float*)d_in[15], (const float*)d_in[16], (float*)d_out, NNODES);
}
